// ObjectRelationModule_2877628089011
// MI455X (gfx1250) — matrix operand order repaired and checked by reading, not compiled, not run on hardware
//
#include <hip/hip_runtime.h>
#include <hip/hip_bf16.h>

// ---------------------------------------------------------------------------
// ObjectRelationModule on MI455X (gfx1250), wave32 + WMMA bf16 16x16x32.
// N=1024 RoIs, 32 heads, d_k=64, d_v=32, d_g=64, D_F=1024.
//
// The reference's global-min shift cancels in the row normalization, so we
// use a per-row max (flash-attention style) computed inside the fused pass.
// Geometry (256 MB, streamed once) uses non-temporal loads so the reused
// working set (q/k/v/g_t, ~80 MB) stays resident in the 192 MB L2.
// ---------------------------------------------------------------------------

typedef __attribute__((ext_vector_type(16))) __bf16 v16bf;
typedef __attribute__((ext_vector_type(8)))  __bf16 v8bf;
typedef __attribute__((ext_vector_type(8)))  float  v8f;

#define NROI   1024
#define NHEAD  32
#define DK     64
#define DV     32
#define DG     64
#define DF     1024

__device__ __forceinline__ v8f wmma_bf16(v16bf a, v16bf b, v8f c) {
  // (neg_a, A, neg_b, B, c_mod, C, reuse_a, reuse_b)
  return __builtin_amdgcn_wmma_f32_16x16x32_bf16(false, a, false, b, (short)0, c,
                                                 false, false);
}

// A-matrix 16x32 bf16 fragment: lane m (<16) holds K={kh*8..kh*8+7, 16+kh*8..}
__device__ __forceinline__ v16bf load_afrag_bf(const __bf16* A, int row_stride,
                                               int r0, int kbase, int lane) {
  int m = lane & 15, kh = lane >> 4;
  const __bf16* p = A + (size_t)(r0 + m) * row_stride + kbase;
  v16bf a;
#pragma unroll
  for (int e = 0; e < 8; ++e) a[e] = p[kh * 8 + e];
#pragma unroll
  for (int e = 0; e < 8; ++e) a[8 + e] = p[16 + kh * 8 + e];
  return a;
}

// B-matrix 32x16 bf16 fragment from row-major W[Nout][K]: B[k][n] = W[n0+n][k].
// lane n holds K = {kh*8..kh*8+7, 16+kh*8..16+kh*8+7}.
__device__ __forceinline__ v16bf load_bfrag_bf(const __bf16* W, int row_stride,
                                               int n0, int kbase, int lane) {
  int n = lane & 15, kh = lane >> 4;
  const __bf16* p = W + (size_t)(n0 + n) * row_stride + kbase;
  v16bf b;
#pragma unroll
  for (int e = 0; e < 8; ++e) b[e] = p[kh * 8 + e];
#pragma unroll
  for (int e = 0; e < 8; ++e) b[8 + e] = p[16 + kh * 8 + e];
  return b;
}

// ---------------------------------------------------------------------------
// f32 -> bf16 elementwise conversion
// ---------------------------------------------------------------------------
__global__ void cvt_f32_bf16(const float* __restrict__ src,
                             __bf16* __restrict__ dst, int n) {
  for (int i = blockIdx.x * blockDim.x + threadIdx.x; i < n;
       i += gridDim.x * blockDim.x)
    dst[i] = (__bf16)src[i];
}

// ---------------------------------------------------------------------------
// Generic GEMM: C[M][Nout] = A[M][K] * W[Nout][K]^T  (all bf16, f32 accum)
// block = 1 wave, computes a 16x64 strip. grid = (M/16, Nout/64)
// ---------------------------------------------------------------------------
__global__ __launch_bounds__(32) void gemm_bf16_nt(const __bf16* __restrict__ A,
                                                   const __bf16* __restrict__ W,
                                                   __bf16* __restrict__ C,
                                                   int Nout, int K) {
  int lane = threadIdx.x & 31;
  int i0 = blockIdx.x * 16;
  int c0 = blockIdx.y * 64;
  v8f acc[4] = {v8f{}, v8f{}, v8f{}, v8f{}};
  for (int ks = 0; ks < K; ks += 32) {
    v16bf a = load_afrag_bf(A, K, i0, ks, lane);
#pragma unroll
    for (int nt = 0; nt < 4; ++nt) {
      v16bf b = load_bfrag_bf(W, K, c0 + nt * 16, ks, lane);
      acc[nt] = wmma_bf16(a, b, acc[nt]);
    }
  }
  int n = lane & 15, hi = lane >> 4;
#pragma unroll
  for (int nt = 0; nt < 4; ++nt)
#pragma unroll
    for (int r = 0; r < 8; ++r)
      C[(size_t)(i0 + r + 8 * hi) * Nout + c0 + nt * 16 + n] =
          (__bf16)acc[nt][r];
}

// ---------------------------------------------------------------------------
// Geometry weights: g_t[h][p] = geom[p,:64] . Wg[h,:] + bg[h]   (bf16 out,
// transposed so the attention pass reads contiguous runs per head).
// Grid-stride over 16-row tiles; one wave per block computes 16x32.
// Geometry is streamed once -> non-temporal loads + next-tile prefetch.
// Per lane the 8 outputs per head-half are contiguous -> packed b128 stores.
// ---------------------------------------------------------------------------
__global__ __launch_bounds__(32) void geo_kernel(const float* __restrict__ geom,
                                                 const __bf16* __restrict__ Wg,
                                                 const float* __restrict__ bg,
                                                 __bf16* __restrict__ g_t) {
  const size_t NN = (size_t)NROI * NROI;
  const int ntiles = (NROI * NROI) / 16;
  int lane = threadIdx.x & 31;
  int m = lane & 15, kh = lane >> 4;
  int n = lane & 15, hi = lane >> 4;
  float bias0 = bg[n], bias1 = bg[16 + n];

  for (int tile = blockIdx.x; tile < ntiles; tile += gridDim.x) {
    int p0 = tile * 16;
    // prefetch this lane's rows of the next tile we will process
    if (tile + gridDim.x < ntiles)
      __builtin_prefetch(geom + (size_t)(p0 + (int)gridDim.x * 16 + m) * DG, 0, 1);

    v8f c0 = {}, c1 = {};
#pragma unroll
    for (int ks = 0; ks < 2; ++ks) {
      const float* pa = geom + (size_t)(p0 + m) * DG + ks * 32;
      v16bf a;
#pragma unroll
      for (int e = 0; e < 8; ++e)
        a[e] = (__bf16)__builtin_nontemporal_load(pa + kh * 8 + e);
#pragma unroll
      for (int e = 0; e < 8; ++e)
        a[8 + e] = (__bf16)__builtin_nontemporal_load(pa + 16 + kh * 8 + e);
      v16bf b0 = load_bfrag_bf(Wg, DG, 0, ks * 32, lane);
      v16bf b1 = load_bfrag_bf(Wg, DG, 16, ks * 32, lane);
      c0 = wmma_bf16(a, b0, c0);
      c1 = wmma_bf16(a, b1, c1);
    }
    // pack the 8 contiguous p-results per lane into one 16B store per head
    v8bf pk0, pk1;
#pragma unroll
    for (int r = 0; r < 8; ++r) {
      pk0[r] = (__bf16)(c0[r] + bias0);
      pk1[r] = (__bf16)(c1[r] + bias1);
    }
    *(v8bf*)(g_t + (size_t)n * NN + p0 + 8 * hi) = pk0;
    *(v8bf*)(g_t + (size_t)(16 + n) * NN + p0 + 8 * hi) = pk1;
  }
}

// ---------------------------------------------------------------------------
// Fused attention: per (16-row i-tile, head). 4 waves / block.
//   phase1: s = q.k^T (WMMA) -> LDS (16x1024 f32)
//   phase2: per-row max, w = exp(s-max)*g, row sums
//   phase3: acc = w @ v (WMMA, w re-fragmented from LDS as bf16)
//   final : out = acc/rowsum + pooled
// ---------------------------------------------------------------------------
__global__ __launch_bounds__(128) void attn_kernel(
    const __bf16* __restrict__ qb, const __bf16* __restrict__ kb,
    const __bf16* __restrict__ vb, const __bf16* __restrict__ g_t,
    const float* __restrict__ pooled, float* __restrict__ out) {
  __shared__ float  s_buf[16 * NROI];      // 64 KB scores -> weights
  __shared__ __bf16 vT[DV * NROI];         // 64 KB V transposed [dv][j]
  __shared__ float  red[16 * 8];
  __shared__ float  rowmaxF[16];
  __shared__ float  rowsumF[16];
  __shared__ float  accred[4][16 * DV];    // 8 KB per-wave partials

  const size_t NN = (size_t)NROI * NROI;
  int t = threadIdx.x;
  int lane = t & 31;
  int wave = t >> 5;
  int i0 = blockIdx.x * 16;
  int h = blockIdx.y;
  int n = lane & 15, hi = lane >> 4;
  int m = lane & 15, kh = lane >> 4;

  // stage V head-slice into LDS transposed: vT[dv][j]
  for (int j = t; j < NROI; j += 128) {
    const __bf16* vp = vb + (size_t)j * DF + h * DV;
#pragma unroll
    for (int dv = 0; dv < DV; ++dv) vT[dv * NROI + j] = vp[dv];
  }

  // q fragments for this i-tile (constant over the j loop)
  v16bf qa0 = load_afrag_bf(qb, NHEAD * DK, i0, h * DK, lane);
  v16bf qa1 = load_afrag_bf(qb, NHEAD * DK, i0, h * DK + 32, lane);

  // ---- phase 1: score tiles -------------------------------------------------
  for (int jt = wave; jt < NROI / 16; jt += 4) {
    int j0 = jt * 16;
    v16bf kf0 = load_bfrag_bf(kb, NHEAD * DK, j0, h * DK, lane);
    v16bf kf1 = load_bfrag_bf(kb, NHEAD * DK, j0, h * DK + 32, lane);
    v8f c = {};
    c = wmma_bf16(qa0, kf0, c);
    c = wmma_bf16(qa1, kf1, c);
#pragma unroll
    for (int r = 0; r < 8; ++r) s_buf[(r + 8 * hi) * NROI + j0 + n] = c[r];
  }
  __syncthreads();

  // ---- phase 2: row max, weights, row sums ---------------------------------
  int row = t >> 3, ch = t & 7;  // 8 threads per row, 128 cols each
  {
    float mx = -3.4e38f;
    const float* sp = s_buf + row * NROI + ch * 128;
    for (int j = 0; j < 128; ++j) mx = fmaxf(mx, sp[j]);
    red[row * 8 + ch] = mx;
  }
  __syncthreads();
  if (ch == 0) {
    float mx = red[row * 8];
    for (int k = 1; k < 8; ++k) mx = fmaxf(mx, red[row * 8 + k]);
    rowmaxF[row] = mx;
  }
  __syncthreads();
  {
    float mx = rowmaxF[row];
    float* sp = s_buf + row * NROI + ch * 128;
    const __bf16* gp = g_t + (size_t)h * NN + (size_t)(i0 + row) * NROI + ch * 128;
    float sum = 0.f;
    for (int j = 0; j < 128; ++j) {
      // g_t element is consumed exactly once across the whole dispatch -> NT
      float w = __expf(sp[j] - mx) * (float)__builtin_nontemporal_load(gp + j);
      sp[j] = w;
      sum += w;
    }
    red[row * 8 + ch] = sum;
  }
  __syncthreads();
  if (ch == 0) {
    float s = 0.f;
    for (int k = 0; k < 8; ++k) s += red[row * 8 + k];
    rowsumF[row] = s;
  }
  __syncthreads();

  // ---- phase 3: acc = w @ v -------------------------------------------------
  v8f acc0 = {}, acc1 = {};
  for (int jc = wave; jc < NROI / 32; jc += 4) {
    int j32 = jc * 32;
    v16bf a;
    const float* wp = s_buf + m * NROI + j32;
#pragma unroll
    for (int e = 0; e < 8; ++e) a[e] = (__bf16)wp[kh * 8 + e];
#pragma unroll
    for (int e = 0; e < 8; ++e) a[8 + e] = (__bf16)wp[16 + kh * 8 + e];
    v16bf b0, b1;
    const __bf16* v0p = vT + n * NROI + j32;
    const __bf16* v1p = vT + (16 + n) * NROI + j32;
#pragma unroll
    for (int e = 0; e < 8; ++e) { b0[e] = v0p[kh * 8 + e]; b1[e] = v1p[kh * 8 + e]; }
#pragma unroll
    for (int e = 0; e < 8; ++e) { b0[8 + e] = v0p[16 + kh * 8 + e]; b1[8 + e] = v1p[16 + kh * 8 + e]; }
    acc0 = wmma_bf16(a, b0, acc0);
    acc1 = wmma_bf16(a, b1, acc1);
  }
#pragma unroll
  for (int r = 0; r < 8; ++r) {
    accred[wave][(r + 8 * hi) * DV + n] = acc0[r];
    accred[wave][(r + 8 * hi) * DV + 16 + n] = acc1[r];
  }
  __syncthreads();

  // ---- final: normalize + residual -----------------------------------------
  for (int idx = t * 4; idx < t * 4 + 4; ++idx) {
    int rr = idx >> 5, dv = idx & 31;
    float v = accred[0][rr * DV + dv] + accred[1][rr * DV + dv] +
              accred[2][rr * DV + dv] + accred[3][rr * DV + dv];
    v /= rowsumF[rr];
    size_t o = (size_t)(i0 + rr) * DF + h * DV + dv;
    out[o] = v + pooled[o];
  }
}

// ---------------------------------------------------------------------------
extern "C" void kernel_launch(void* const* d_in, const int* in_sizes, int n_in,
                              void* d_out, int out_size, void* d_ws,
                              size_t ws_size, hipStream_t stream) {
  const float* pooled = (const float*)d_in[1];
  const float* geom   = (const float*)d_in[3];  // [N*N, 64, 1, 1]
  const float* Wq     = (const float*)d_in[4];  // [2048, 1024]
  const float* Wk     = (const float*)d_in[5];
  const float* Wv     = (const float*)d_in[6];  // [1024, 1024]
  const float* Wg     = (const float*)d_in[7];  // [32, 64]
  const float* bg     = (const float*)d_in[8];  // [32]
  float* out = (float*)d_out;

  // workspace carve-up (~86 MB total)
  char* ws = (char*)d_ws;
  size_t off = 0;
  auto take = [&](size_t bytes) { char* p = ws + off; off += (bytes + 255) & ~(size_t)255; return p; };
  __bf16* pooled_bf = (__bf16*)take((size_t)NROI * DF * 2);
  __bf16* Wq_bf     = (__bf16*)take((size_t)NHEAD * DK * DF * 2);
  __bf16* Wk_bf     = (__bf16*)take((size_t)NHEAD * DK * DF * 2);
  __bf16* Wv_bf     = (__bf16*)take((size_t)DF * DF * 2);
  __bf16* Wg_bf     = (__bf16*)take((size_t)NHEAD * DG * 2);
  __bf16* q_bf      = (__bf16*)take((size_t)NROI * NHEAD * DK * 2);
  __bf16* k_bf      = (__bf16*)take((size_t)NROI * NHEAD * DK * 2);
  __bf16* v_bf      = (__bf16*)take((size_t)NROI * DF * 2);
  __bf16* g_t       = (__bf16*)take((size_t)NHEAD * NROI * NROI * 2);

  // conversions
  cvt_f32_bf16<<<1024, 256, 0, stream>>>(pooled, pooled_bf, NROI * DF);
  cvt_f32_bf16<<<2048, 256, 0, stream>>>(Wq, Wq_bf, NHEAD * DK * DF);
  cvt_f32_bf16<<<2048, 256, 0, stream>>>(Wk, Wk_bf, NHEAD * DK * DF);
  cvt_f32_bf16<<<1024, 256, 0, stream>>>(Wv, Wv_bf, DF * DF);
  cvt_f32_bf16<<<1, 256, 0, stream>>>(Wg, Wg_bf, NHEAD * DG);

  // projections (WMMA bf16)
  gemm_bf16_nt<<<dim3(NROI / 16, (NHEAD * DK) / 64), 32, 0, stream>>>(
      pooled_bf, Wq_bf, q_bf, NHEAD * DK, DF);
  gemm_bf16_nt<<<dim3(NROI / 16, (NHEAD * DK) / 64), 32, 0, stream>>>(
      pooled_bf, Wk_bf, k_bf, NHEAD * DK, DF);
  gemm_bf16_nt<<<dim3(NROI / 16, DF / 64), 32, 0, stream>>>(
      pooled_bf, Wv_bf, v_bf, DF, DF);

  // geometry weights, transposed per head (grid-stride, 4 tiles/block)
  geo_kernel<<<16384, 32, 0, stream>>>(geom, Wg_bf, bg, g_t);

  // fused attention + residual
  attn_kernel<<<dim3(NROI / 16, NHEAD), 128, 0, stream>>>(q_bf, k_bf, v_bf, g_t,
                                                          pooled, out);
}
